// GNNCritic4_1_27195732918305
// MI455X (gfx1250) — matrix operand order repaired and checked by reading, not compiled, not run on hardware
//
#include <hip/hip_runtime.h>
#include <hip/hip_bf16.h>

#define BATCH 1024
#define NREG  16
#define CDIM  128
#define HDIM  256
#define NNODE (BATCH*NREG)      // 16384
#define NEDGE 131072
#define NE    (NREG*NREG)       // 256

typedef __attribute__((ext_vector_type(16))) __bf16        v16bf;
typedef __attribute__((ext_vector_type(8)))  float         v8f;
typedef __attribute__((ext_vector_type(4)))  unsigned int  v4u;
typedef __attribute__((ext_vector_type(8)))  int           v8i;
typedef __attribute__((ext_vector_type(4)))  int           v4i;

#define WMMA_BF16(a, b, c) \
    __builtin_amdgcn_wmma_f32_16x16x32_bf16(false, (a), false, (b), (short)0, (c), false, false)

__device__ __forceinline__ __bf16 tobf(float f) {
    __hip_bfloat16 h = __float2bfloat16(f);
    return __builtin_bit_cast(__bf16, h);
}

// load 16 contiguous bf16 (32 bytes, 16B-aligned) as a WMMA operand vector
__device__ __forceinline__ v16bf load16bf(const __bf16* p) {
    union { uint4 u[2]; v16bf v; } t;
    const uint4* q = reinterpret_cast<const uint4*>(p);
    t.u[0] = q[0];
    t.u[1] = q[1];
    return t.v;
}

// permute K-within-32 into A-operand storage order (groups of 8: [0,2,1,3])
__device__ __forceinline__ int aperm(int kl) {
    int g  = kl >> 3;
    int ng = (0x3120 >> (g * 4)) & 0xF;
    return ng * 8 + (kl & 7);
}

// LDS byte offset of a __shared__ object (generic addr low 32 bits = LDS addr)
__device__ __forceinline__ unsigned lds_off(const void* p) {
    return (unsigned)(uintptr_t)p;
}

// ---------------------------------------------------------------- TDM load
// 2D tile (rows x d0 elems of bf16), row stride stride0 elems in memory,
// LDS destination padded by (pad_amt+1) dwords every 2^(pad_int+1) dwords.
__device__ __forceinline__ void tdm_load_2d(unsigned lds_addr, const void* gsrc,
                                            unsigned d0, unsigned rows,
                                            unsigned stride0,
                                            unsigned pad_int, unsigned pad_amt) {
    unsigned long long ga = (unsigned long long)(uintptr_t)gsrc;
    v4u g0;
    g0.x = 1u;                                               // count=1 (valid user D#)
    g0.y = lds_addr;                                         // LDS byte address
    g0.z = (unsigned)(ga & 0xFFFFFFFFu);                     // global_addr[31:0]
    g0.w = (unsigned)((ga >> 32) & 0x1FFFFFFu) | (2u << 30); // global_addr[56:32] | type=2
    v8i g1;
    g1[0] = (int)((1u << 16)                 // data_size = 2 bytes (bf16)
                | (1u << 20)                 // pad_enable
                | (pad_int << 22)            // pad every 2^(pad_int+1) dwords
                | (pad_amt << 25));          // pad (pad_amt+1) dwords
    g1[1] = (int)((d0 & 0xFFFFu) << 16);                     // tensor_dim0[15:0]
    g1[2] = (int)(((d0 >> 16) & 0xFFFFu) | ((rows & 0xFFFFu) << 16)); // dim0 hi | dim1 lo
    g1[3] = (int)(((rows >> 16) & 0xFFFFu) | ((d0 & 0xFFFFu) << 16)); // dim1 hi | tile_dim0
    g1[4] = (int)(rows & 0xFFFFu);                           // tile_dim1 (tile_dim2=0)
    g1[5] = (int)stride0;                                    // tensor_dim0_stride lo32
    g1[6] = 0;
    g1[7] = 0;
    v4i z4 = {0, 0, 0, 0};
#if __clang_major__ >= 23
    v8i z8 = {0, 0, 0, 0, 0, 0, 0, 0};
    __builtin_amdgcn_tensor_load_to_lds(g0, g1, z4, z4, z8, 0);
#else
    __builtin_amdgcn_tensor_load_to_lds(g0, g1, z4, z4, 0);
#endif
}

// ---------------------------------------------------------------- utilities
__global__ void fill_f32(float* p, float v, int n) {
    int i = blockIdx.x * blockDim.x + threadIdx.x;
    if (i < n) p[i] = v;
}

__global__ void cvt_bf16(const float* __restrict__ s, __bf16* __restrict__ d, int n) {
    int i = blockIdx.x * blockDim.x + threadIdx.x;
    if (i < n) d[i] = tobf(s[i]);
}

__global__ void deg_kernel(const long long* __restrict__ ei, float* __restrict__ deg, int e) {
    int i = blockIdx.x * blockDim.x + threadIdx.x;
    if (i < e) atomicAdd(&deg[(int)ei[e + i]], 1.0f);
}

__global__ void rsqrt_kernel(const float* __restrict__ deg, float* __restrict__ dinv, int n) {
    int i = blockIdx.x * blockDim.x + threadIdx.x;
    if (i < n) dinv[i] = rsqrtf(deg[i]);   // deg >= 1 (self loop)
}

__global__ __launch_bounds__(256)
void scatter_kernel(const long long* __restrict__ ei, const float* __restrict__ xw,
                    const float* __restrict__ dinv, float* __restrict__ agg, int e) {
    int edge = blockIdx.x * 2 + (threadIdx.x >> 7);
    int c    = threadIdx.x & 127;
    int s = (int)ei[edge];
    int d = (int)ei[e + edge];
    float norm = dinv[s] * dinv[d];
    atomicAdd(&agg[d * CDIM + c], xw[s * CDIM + c] * norm);
}

__global__ void combine_kernel(const float* __restrict__ agg, const float* __restrict__ xw,
                               const float* __restrict__ dinv, const float* __restrict__ bc,
                               const float* __restrict__ state, float* __restrict__ x, int n) {
    int i = blockIdx.x * blockDim.x + threadIdx.x;
    if (i >= n) return;
    int row = i >> 7, c = i & 127;
    float di = dinv[row];
    float v  = agg[i] + xw[i] * di * di + bc[c];
    x[i] = fmaxf(v, 0.0f) + state[i];
}

// ------------------------------------------------- GCN GEMM: xw = state @ Wc
// M=16384, N=128, K=128.  32 rows/block.  Wc is TDM-loaded once into LDS with
// a 4-dword row pad (stride 136 bf16) for conflict-free B fetches.
__global__ __launch_bounds__(256)
void gcn_gemm(const float* __restrict__ state, const __bf16* __restrict__ wcb,
              float* __restrict__ xw) {
    __shared__ __attribute__((aligned(16))) __bf16 as[32 * 136];
    __shared__ __attribute__((aligned(16))) __bf16 wcs[128 * 136];
    int tid  = threadIdx.x;
    int lane = tid & 31, wave = tid >> 5;
    int half = lane >> 4, lm = lane & 15;
    int m0   = blockIdx.x * 32;

    if (wave == 0)      // TDM: 128x128 bf16, pad 4 dwords every 64 dwords (one row)
        tdm_load_2d(lds_off(wcs), wcb, CDIM, CDIM, CDIM, /*pad_int=*/5, /*pad_amt=*/3);

    for (int j = tid; j < 32 * 128; j += 256) {      // overlap with the DMA
        int r = j >> 7, c = j & 127;
        as[r * 136 + (c >> 5) * 32 + aperm(c & 31)] = tobf(state[(m0 + r) * CDIM + c]);
    }
    if (wave == 0) __builtin_amdgcn_s_wait_tensorcnt(0);
    __syncthreads();

    int nt = wave;                                   // wave owns one N column
    v8f acc[2];
    acc[0] = {}; acc[1] = {};
#pragma unroll
    for (int kk = 0; kk < 4; ++kk) {
        v16bf bm;
#pragma unroll
        for (int s = 0; s < 16; ++s)
            bm[s] = wcs[(kk * 32 + (s >> 3) * 16 + half * 8 + (s & 7)) * 136 + nt * 16 + lm];
#pragma unroll
        for (int mt = 0; mt < 2; ++mt) {
            v16bf a = load16bf(&as[(mt * 16 + lm) * 136 + kk * 32 + half * 16]);
            acc[mt] = WMMA_BF16(a, bm, acc[mt]);
        }
    }
#pragma unroll
    for (int mt = 0; mt < 2; ++mt) {
        int mbase = m0 + mt * 16 + half * 8;
        int n = nt * 16 + lm;
#pragma unroll
        for (int r = 0; r < 8; ++r)
            xw[(mbase + r) * CDIM + n] = acc[mt][r];
    }
}

// ------------------------------------------- fused per-batch OD-pair MLP head
// One block per batch element.  W1/W2 K-panels (32x256 bf16, 16KB) stream in
// via double-buffered TDM loads (padded to 264-elem rows) while WMMAs run on
// the previous panel.  Wave w owns N columns {2w,2w+1} x 4 M-tiles.
__global__ __launch_bounds__(256)
void mlp_kernel(const float* __restrict__ xbuf, const float* __restrict__ action,
                const long long* __restrict__ edges,
                const __bf16* __restrict__ w1b, const float* __restrict__ W1,
                const float* __restrict__ b1,  const __bf16* __restrict__ w2b,
                const float* __restrict__ b2,  const float* __restrict__ W3,
                const float* __restrict__ b3,  float* __restrict__ out) {
    __shared__ __attribute__((aligned(16))) __bf16 xs[16 * 136];
    __shared__ __attribute__((aligned(16))) __bf16 h1[64 * 264];
    __shared__ __attribute__((aligned(16))) __bf16 pan[2][32 * 264];
    __shared__ int    ep[NE][2];
    __shared__ float  acts[NE];
    __shared__ float  red[HDIM];

    int b = blockIdx.x, tid = threadIdx.x;
    for (int j = tid; j < 16 * 128; j += 256) {
        int r = j >> 7, c = j & 127;
        xs[r * 136 + (c >> 5) * 32 + aperm(c & 31)] = tobf(xbuf[(b * NREG + r) * CDIM + c]);
    }
    ep[tid][0] = (int)edges[tid * 2 + 0];
    ep[tid][1] = (int)edges[tid * 2 + 1];
    acts[tid]  = action[b * NE + tid];
    red[tid]   = 0.0f;
    __syncthreads();

    int lane = tid & 31, wave = tid >> 5;
    int half = lane >> 4, lm = lane & 15;
    int nt0  = wave * 2;
    int n0 = nt0 * 16 + lm, n1 = n0 + 16;

    for (int p = 0; p < 4; ++p) {                     // 4 slabs of 64 OD-pairs
        // =========== GEMM1: h1 = relu(cat @ W1 + b1) (this slab) ===========
        v8f acc[2][4];
        {
            float w1r0 = W1[256 * HDIM + n0], bb0 = b1[n0];
            float w1r1 = W1[256 * HDIM + n1], bb1 = b1[n1];
#pragma unroll
            for (int mt = 0; mt < 4; ++mt)
#pragma unroll
                for (int r = 0; r < 8; ++r) {
                    float av = acts[p * 64 + mt * 16 + half * 8 + r];
                    acc[0][mt][r] = av * w1r0 + bb0;  // action rank-1 + bias init
                    acc[1][mt][r] = av * w1r1 + bb1;
                }
        }
        if (wave == 0) {
            tdm_load_2d(lds_off(pan[0]), w1b, HDIM, 32, HDIM, 6, 3);
            __builtin_amdgcn_s_wait_tensorcnt(0);
        }
        __syncthreads();
        for (int kk = 0; kk < 8; ++kk) {
            if (wave == 0 && kk + 1 < 8)              // prefetch next panel (overlapped)
                tdm_load_2d(lds_off(pan[(kk + 1) & 1]), w1b + (kk + 1) * 32 * HDIM,
                            HDIM, 32, HDIM, 6, 3);
            const __bf16* pb = pan[kk & 1];
            v16bf bm0, bm1;
#pragma unroll
            for (int s = 0; s < 16; ++s) {
                int kr = (s >> 3) * 16 + half * 8 + (s & 7);
                bm0[s] = pb[kr * 264 + nt0 * 16 + lm];
                bm1[s] = pb[kr * 264 + nt0 * 16 + 16 + lm];
            }
#pragma unroll
            for (int mt = 0; mt < 4; ++mt) {
                int node = ep[p * 64 + mt * 16 + lm][kk >> 2];  // src / dst half
                v16bf a  = load16bf(&xs[node * 136 + (kk & 3) * 32 + half * 16]);
                acc[0][mt] = WMMA_BF16(a, bm0, acc[0][mt]);
                acc[1][mt] = WMMA_BF16(a, bm1, acc[1][mt]);
            }
            if (wave == 0 && kk + 1 < 8) __builtin_amdgcn_s_wait_tensorcnt(0);
            __syncthreads();
        }
#pragma unroll
        for (int j = 0; j < 2; ++j) {
            int n   = nt0 * 16 + j * 16 + lm;
            int pos = (n >> 5) * 32 + aperm(n & 31);
#pragma unroll
            for (int mt = 0; mt < 4; ++mt)
#pragma unroll
                for (int r = 0; r < 8; ++r) {
                    int ml = mt * 16 + half * 8 + r;
                    h1[ml * 264 + pos] = tobf(fmaxf(acc[j][mt][r], 0.0f));
                }
        }
        __syncthreads();

        // ====== GEMM2: colsum += sum_m relu(h1 @ W2 + b2) (this slab) ======
        {
            float bb0 = b2[n0], bb1 = b2[n1];
#pragma unroll
            for (int mt = 0; mt < 4; ++mt)
#pragma unroll
                for (int r = 0; r < 8; ++r) {
                    acc[0][mt][r] = bb0;
                    acc[1][mt][r] = bb1;
                }
        }
        if (wave == 0) {
            tdm_load_2d(lds_off(pan[0]), w2b, HDIM, 32, HDIM, 6, 3);
            __builtin_amdgcn_s_wait_tensorcnt(0);
        }
        __syncthreads();
        for (int kk = 0; kk < 8; ++kk) {
            if (wave == 0 && kk + 1 < 8)
                tdm_load_2d(lds_off(pan[(kk + 1) & 1]), w2b + (kk + 1) * 32 * HDIM,
                            HDIM, 32, HDIM, 6, 3);
            const __bf16* pb = pan[kk & 1];
            v16bf bm0, bm1;
#pragma unroll
            for (int s = 0; s < 16; ++s) {
                int kr = (s >> 3) * 16 + half * 8 + (s & 7);
                bm0[s] = pb[kr * 264 + nt0 * 16 + lm];
                bm1[s] = pb[kr * 264 + nt0 * 16 + 16 + lm];
            }
#pragma unroll
            for (int mt = 0; mt < 4; ++mt) {
                v16bf a = load16bf(&h1[(mt * 16 + lm) * 264 + kk * 32 + half * 16]);
                acc[0][mt] = WMMA_BF16(a, bm0, acc[0][mt]);
                acc[1][mt] = WMMA_BF16(a, bm1, acc[1][mt]);
            }
            if (wave == 0 && kk + 1 < 8) __builtin_amdgcn_s_wait_tensorcnt(0);
            __syncthreads();
        }
        float cs0 = 0.0f, cs1 = 0.0f;
#pragma unroll
        for (int mt = 0; mt < 4; ++mt)
#pragma unroll
            for (int r = 0; r < 8; ++r) {
                cs0 += fmaxf(acc[0][mt][r], 0.0f);
                cs1 += fmaxf(acc[1][mt][r], 0.0f);
            }
        atomicAdd(&red[n0], cs0);                     // ds_add_f32
        atomicAdd(&red[n1], cs1);
        __syncthreads();
    }

    // --------- out[b] = colsum . W3 + b3 ---------
    float v = red[tid] * W3[tid];
    __syncthreads();
    red[tid] = v;
    __syncthreads();
    for (int s = 128; s > 0; s >>= 1) {
        if (tid < s) red[tid] += red[tid + s];
        __syncthreads();
    }
    if (tid == 0) out[b] = red[0] + b3[0];
}

// ------------------------------------------------------------------- launch
extern "C" void kernel_launch(void* const* d_in, const int* in_sizes, int n_in,
                              void* d_out, int out_size, void* d_ws, size_t ws_size,
                              hipStream_t stream) {
    (void)in_sizes; (void)n_in; (void)out_size; (void)ws_size;
    const float*     state      = (const float*)d_in[0];
    const long long* edge_index = (const long long*)d_in[1];
    const float*     action     = (const float*)d_in[2];
    const long long* edges      = (const long long*)d_in[3];
    const float*     Wc         = (const float*)d_in[4];
    const float*     bc         = (const float*)d_in[5];
    const float*     W1         = (const float*)d_in[6];
    const float*     b1         = (const float*)d_in[7];
    const float*     W2         = (const float*)d_in[8];
    const float*     b2         = (const float*)d_in[9];
    const float*     W3         = (const float*)d_in[10];
    const float*     b3         = (const float*)d_in[11];
    float*           out        = (float*)d_out;

    char* ws = (char*)d_ws;
    size_t off = 0;
    __bf16* wcb  = (__bf16*)(ws + off); off += (size_t)CDIM * CDIM * 2;
    __bf16* w1b  = (__bf16*)(ws + off); off += (size_t)256 * HDIM * 2;
    __bf16* w2b  = (__bf16*)(ws + off); off += (size_t)HDIM * HDIM * 2;
    float*  xw   = (float*)(ws + off);  off += (size_t)NNODE * CDIM * 4;
    float*  agg  = (float*)(ws + off);  off += (size_t)NNODE * CDIM * 4;
    float*  deg  = (float*)(ws + off);  off += (size_t)NNODE * 4;
    float*  dinv = (float*)(ws + off);  off += (size_t)NNODE * 4;
    float*  xv   = (float*)(ws + off);  off += (size_t)NNODE * CDIM * 4;

    cvt_bf16<<<64,  256, 0, stream>>>(Wc, wcb, CDIM * CDIM);
    cvt_bf16<<<256, 256, 0, stream>>>(W1, w1b, 256 * HDIM);      // rows 0..255 only
    cvt_bf16<<<256, 256, 0, stream>>>(W2, w2b, HDIM * HDIM);

    fill_f32<<<64,   256, 0, stream>>>(deg, 1.0f, NNODE);        // self-loop degree
    fill_f32<<<8192, 256, 0, stream>>>(agg, 0.0f, NNODE * CDIM);

    deg_kernel<<<NEDGE / 256, 256, 0, stream>>>(edge_index, deg, NEDGE);
    rsqrt_kernel<<<64, 256, 0, stream>>>(deg, dinv, NNODE);

    gcn_gemm<<<NNODE / 32, 256, 0, stream>>>(state, wcb, xw);

    scatter_kernel<<<NEDGE / 2, 256, 0, stream>>>(edge_index, xw, dinv, agg, NEDGE);
    combine_kernel<<<(NNODE * CDIM) / 256, 256, 0, stream>>>(agg, xw, dinv, bc, state, xv,
                                                             NNODE * CDIM);

    mlp_kernel<<<BATCH, 256, 0, stream>>>(xv, action, edges, w1b, W1, b1,
                                          w2b, b2, W3, b3, out);
}
